// GraphNN_85572928405520
// MI455X (gfx1250) — matrix operand order repaired and checked by reading, not compiled, not run on hardware
//
#include <hip/hip_runtime.h>
#include <hip/hip_bf16.h>
#include <math.h>

typedef __attribute__((ext_vector_type(16))) _Float16 v16h;
typedef __attribute__((ext_vector_type(8)))  _Float16 v8h;
typedef __attribute__((ext_vector_type(4)))  _Float16 v4h;
typedef __attribute__((ext_vector_type(8)))  float    v8f;

// ---------- helpers: order-preserving float <-> uint encoding for atomicMax ----------
static __device__ __forceinline__ unsigned enc_f32(float f) {
  unsigned u = __float_as_uint(f);
  return (u & 0x80000000u) ? ~u : (u | 0x80000000u);
}
static __device__ __forceinline__ float dec_f32(unsigned e) {
  return __uint_as_float((e & 0x80000000u) ? (e ^ 0x80000000u) : ~e);
}
#define ENC_NEG_INF 0x007FFFFFu  // enc(-inf)

// =====================================================================
// WMMA GEMM:  Y[N x M] = act( X[N x K](f32) * W[K x M] + bias )
// WT is W^T stored [M x K] in fp16.
// Block = 8 waves covering 128 output columns x 16 rows.
// The 16 x K A-tile is staged ONCE per block into LDS as f16
// (cooperative float4 loads + packed converts), then each wave reads
// its WMMA A-fragments with 16B ds loads -> 8x less A-side global
// traffic + conversion work vs per-wave loading.
// A-fragment lane layout per CDNA5 ISA 7.12.2 (16-bit A 16x32):
//   lane L: row M = L%16, half = L/16; elems 0..7  -> K = k0+half*8+i,
//   elems 8..15 -> K = k0+16+half*8+(i-8)   (two contiguous 16B spans)
// B-fragment: lane L: col N = L%16, same K mapping as A-fragment
// C/D: vgpr r, lane L: (M = r + 8*(L/16), N = L%16)
// =====================================================================
__global__ __launch_bounds__(256)
void k_gemm_wmma(const float* __restrict__ X,
                 const _Float16* __restrict__ WT,   // [M x K] fp16 (pre-transposed)
                 const float* __restrict__ bias,    // [M] or nullptr
                 float* __restrict__ Y,
                 int K, int M, int relu)
{
  __shared__ _Float16 smA[16 * 256];                // max K = 256 -> 8 KB
  const int lane = threadIdx.x & 31;
  const int wave = threadIdx.x >> 5;
  const int half = lane >> 4;
  const int mrow = lane & 15;
  const int rowBase = blockIdx.y * 16;
  const int colBase = (blockIdx.x * 8 + wave) * 16;
  const int col = colBase + mrow;

  // ---- stage A tile (16 rows x K) as f16 into LDS, once ----
  for (int i = threadIdx.x * 4; i < 16 * K; i += 256 * 4) {
    const int r = i / K, cc = i % K;
    float4 f = *(const float4*)(X + (size_t)(rowBase + r) * K + cc);
    v4h h;
    h[0] = (_Float16)f.x; h[1] = (_Float16)f.y;
    h[2] = (_Float16)f.z; h[3] = (_Float16)f.w;
    *(v4h*)(&smA[r * K + cc]) = h;
  }
  __syncthreads();

  v8f c = {};
  const _Float16* arow = &smA[mrow * K];
  const _Float16* bp0  = WT + (size_t)col * K + half * 8;

  for (int k0 = 0; k0 < K; k0 += 32) {
    const _Float16* ap = arow + k0 + half * 8;
    v8h alo = *(const v8h*)ap;                     // ds_load_b128
    v8h ahi = *(const v8h*)(ap + 16);              // ds_load_b128
    const _Float16* bp = bp0 + k0;
    v8h blo = *(const v8h*)bp;                     // global_load_b128
    v8h bhi = *(const v8h*)(bp + 16);              // global_load_b128
    v16h a, b;
#pragma unroll
    for (int i = 0; i < 8; ++i) {
      a[i] = alo[i]; a[8 + i] = ahi[i];
      b[i] = blo[i]; b[8 + i] = bhi[i];
    }
    c = __builtin_amdgcn_wmma_f32_16x16x32_f16(
        /*neg_a=*/false, a, /*neg_b=*/false, b,
        /*c_mod=*/(short)0, c, /*reuse_a=*/false, /*reuse_b=*/false);
  }

  const float bv = bias ? bias[col] : 0.0f;
#pragma unroll
  for (int r = 0; r < 8; ++r) {
    const int row = rowBase + r + 8 * half;
    float v = c[r] + bv;
    if (relu) v = fmaxf(v, 0.0f);
    Y[(size_t)row * M + col] = v;
  }
}

// ---------- weight convert + transpose: WT[m*K + k] = (f16) W[k*ld + m] ----------
__global__ void k_convert_T(const float* __restrict__ W, _Float16* __restrict__ WT,
                            int K, int M, int ld)
{
  int i = blockIdx.x * blockDim.x + threadIdx.x;
  if (i >= K * M) return;
  int k = i / M, m = i % M;
  WT[(size_t)m * K + k] = (_Float16)W[(size_t)k * ld + m];
}

// ---------- tiny global MLP: g = (gx @ wg1 + bg1) @ wg2 + bg2 ----------
__global__ void k_global_mlp(const float* gx, const float* wg1, const float* bg1,
                             const float* wg2, const float* bg2, float* g)
{
  __shared__ float t[64];
  int tid = threadIdx.x;
  if (tid < 64) {
    float a = bg1[tid];
    for (int i = 0; i < 32; ++i) a += gx[i] * wg1[i * 64 + tid];
    t[tid] = a;
  }
  __syncthreads();
  if (tid < 128) {
    float a = bg2[tid];
    for (int i = 0; i < 64; ++i) a += t[i] * wg2[i * 128 + tid];
    g[tid] = a;
  }
}

// gw1[m] = b1[m] + sum_k g[k] * w1[(128+k)*256 + m]   (bias for head GEMM 1)
__global__ void k_gw1(const float* g, const float* w1, const float* b1, float* gw1)
{
  int m = threadIdx.x;  // 256 threads
  float a = b1[m];
  for (int k = 0; k < 128; ++k) a += g[k] * w1[(size_t)(128 + k) * 256 + m];
  gw1[m] = a;
}

__global__ void k_fill_u32(unsigned* p, unsigned v, int n)
{
  int i = blockIdx.x * blockDim.x + threadIdx.x;
  if (i < n) p[i] = v;
}

// ---------- GATv2 edge pipeline (wave per edge, lane-parallel features) ----------
__global__ __launch_bounds__(256)
void k_edge_scores(const int* __restrict__ ei, int E, int Etot,
                   const float* __restrict__ XL, const float* __restrict__ XR,
                   const float* __restrict__ att,
                   float* __restrict__ eOut, unsigned* __restrict__ segmax, int F)
{
  int gid = blockIdx.x * 8 + (threadIdx.x >> 5);
  if (gid >= Etot) return;
  int lane = threadIdx.x & 31;
  int s, d;
  if (gid < E) { s = ei[gid]; d = ei[E + gid]; } else { s = d = gid - E; }
  const float* pl = XL + (size_t)s * F;
  const float* pr = XR + (size_t)d * F;
  float acc = 0.f;
  for (int k = lane; k < F; k += 32) {
    float v = pl[k] + pr[k];
    v = (v > 0.f) ? v : 0.2f * v;       // leaky_relu slope 0.2
    acc += v * att[k];
  }
#pragma unroll
  for (int off = 16; off > 0; off >>= 1) acc += __shfl_down(acc, off, 32);
  if (lane == 0) {
    eOut[gid] = acc;
    atomicMax(&segmax[d], enc_f32(acc));
  }
}

__global__ void k_edge_exp(const int* __restrict__ ei, int E, int Etot,
                           float* __restrict__ eBuf,
                           const unsigned* __restrict__ segmax,
                           float* __restrict__ segsum)
{
  int gid = blockIdx.x * blockDim.x + threadIdx.x;
  if (gid >= Etot) return;
  int d = (gid < E) ? ei[E + gid] : (gid - E);
  float ee = expf(eBuf[gid] - dec_f32(segmax[d]));
  eBuf[gid] = ee;
  atomicAdd(&segsum[d], ee);
}

__global__ __launch_bounds__(256)
void k_edge_agg(const int* __restrict__ ei, int E, int Etot,
                const float* __restrict__ eExp, const float* __restrict__ segsum,
                const float* __restrict__ XL, float* __restrict__ agg, int F)
{
  int gid = blockIdx.x * 8 + (threadIdx.x >> 5);
  if (gid >= Etot) return;
  int lane = threadIdx.x & 31;
  int s, d;
  if (gid < E) { s = ei[gid]; d = ei[E + gid]; } else { s = d = gid - E; }
  float alpha = eExp[gid] / segsum[d];
  const float* pl = XL + (size_t)s * F;
  float* pd = agg + (size_t)d * F;
  for (int k = lane; k < F; k += 32) atomicAdd(&pd[k], alpha * pl[k]);
}

// ---------- elementwise bias + relu ----------
__global__ void k_bias_relu(float* x, const float* bias, long long n, int M)
{
  long long i = (long long)blockIdx.x * blockDim.x + threadIdx.x;
  if (i >= n) return;
  x[i] = fmaxf(x[i] + bias[i % M], 0.0f);
}

// ---------- BatchNorm (training-mode, biased variance) ----------
__global__ void k_bn_stats(const float* __restrict__ x, float* __restrict__ stats,
                           int N, int M)
{
  int f = threadIdx.x;  // blockDim.x == M
  float s = 0.f, q = 0.f;
  for (int r = blockIdx.x; r < N; r += gridDim.x) {
    float v = x[(size_t)r * M + f];
    s += v; q += v * v;
  }
  atomicAdd(&stats[f], s);
  atomicAdd(&stats[M + f], q);
}

__global__ void k_bn_apply(float* x, const float* stats, const float* g,
                           const float* b, long long n, int N, int M)
{
  long long i = (long long)blockIdx.x * blockDim.x + threadIdx.x;
  if (i >= n) return;
  int f = (int)(i % M);
  float mean = stats[f] / N;
  float var  = stats[M + f] / N - mean * mean;
  x[i] = (x[i] - mean) * rsqrtf(var + 1e-5f) * g[f] + b[f];
}

// ---------- final column (M=1): scores = relu(x @ w3 + b3), wave per row ----------
__global__ __launch_bounds__(256)
void k_rowdot_relu(const float* __restrict__ X, const float* __restrict__ w,
                   const float* __restrict__ b, float* __restrict__ out, int N, int K)
{
  int row = blockIdx.x * 8 + (threadIdx.x >> 5);
  if (row >= N) return;
  int lane = threadIdx.x & 31;
  const float* p = X + (size_t)row * K;
  float acc = 0.f;
  for (int k = lane; k < K; k += 32) acc += p[k] * w[k];
#pragma unroll
  for (int off = 16; off > 0; off >>= 1) acc += __shfl_down(acc, off, 32);
  if (lane == 0) out[row] = fmaxf(acc + b[0], 0.f);
}

// ---------- softmax over all N scores ----------
__global__ void k_softmax_reduce(const float* __restrict__ s, float* sm, int N)
{
  __shared__ float red[1024];
  int tid = threadIdx.x;
  float m = -INFINITY;
  for (int i = tid; i < N; i += 1024) m = fmaxf(m, s[i]);
  red[tid] = m; __syncthreads();
  for (int off = 512; off > 0; off >>= 1) {
    if (tid < off) red[tid] = fmaxf(red[tid], red[tid + off]);
    __syncthreads();
  }
  float mx = red[0]; __syncthreads();
  float acc = 0.f;
  for (int i = tid; i < N; i += 1024) acc += expf(s[i] - mx);
  red[tid] = acc; __syncthreads();
  for (int off = 512; off > 0; off >>= 1) {
    if (tid < off) red[tid] += red[tid + off];
    __syncthreads();
  }
  if (tid == 0) { sm[0] = mx; sm[1] = red[0]; }
}

__global__ void k_softmax_write(const float* __restrict__ s, const float* sm,
                                float* __restrict__ out, int N)
{
  int i = blockIdx.x * blockDim.x + threadIdx.x;
  if (i < N) out[i] = expf(s[i] - sm[0]) / sm[1];
}

// =====================================================================
extern "C" void kernel_launch(void* const* d_in, const int* in_sizes, int n_in,
                              void* d_out, int out_size, void* d_ws, size_t ws_size,
                              hipStream_t stream)
{
  const float* node  = (const float*)d_in[0];
  const int*   ei    = (const int*)  d_in[1];
  const float* gx    = (const float*)d_in[2];
  const float* wg1   = (const float*)d_in[3];
  const float* bg1   = (const float*)d_in[4];
  const float* wg2   = (const float*)d_in[5];
  const float* bg2   = (const float*)d_in[6];
  const float* c1_wl = (const float*)d_in[7];
  const float* c1_wr = (const float*)d_in[8];
  const float* c1_at = (const float*)d_in[9];
  const float* c1_b  = (const float*)d_in[10];
  const float* bn1_g = (const float*)d_in[11];
  const float* bn1_b = (const float*)d_in[12];
  const float* c2_wl = (const float*)d_in[13];
  const float* c2_wr = (const float*)d_in[14];
  const float* c2_at = (const float*)d_in[15];
  const float* c2_b  = (const float*)d_in[16];
  const float* bn2_g = (const float*)d_in[17];
  const float* bn2_b = (const float*)d_in[18];
  const float* w1    = (const float*)d_in[19];
  const float* b1    = (const float*)d_in[20];
  const float* fbn1g = (const float*)d_in[21];
  const float* fbn1b = (const float*)d_in[22];
  const float* w2    = (const float*)d_in[23];
  const float* b2    = (const float*)d_in[24];
  const float* fbn2g = (const float*)d_in[25];
  const float* fbn2b = (const float*)d_in[26];
  const float* w3    = (const float*)d_in[27];
  const float* b3    = (const float*)d_in[28];

  const int F = 128, H = 128;                  // feature / hidden dims
  const int N    = in_sizes[0] / F;            // 30000 (multiple of 16)
  const int E    = in_sizes[1] / 2;            // 480000
  const int Etot = E + N;                      // + self loops

  // ---- workspace carve-up ----
  char* base = (char*)d_ws; size_t off = 0;
  auto alloc = [&](size_t bytes) -> void* {
    void* p = base + off;
    off = (off + bytes + 255) & ~(size_t)255;
    return p;
  };
  float*    gvec  = (float*)alloc(128 * 4);
  float*    gw1v  = (float*)alloc(256 * 4);
  _Float16* wlT1  = (_Float16*)alloc((size_t)128 * 128 * 2);
  _Float16* wrT1  = (_Float16*)alloc((size_t)128 * 128 * 2);
  _Float16* wlT2  = (_Float16*)alloc((size_t)128 * 128 * 2);
  _Float16* wrT2  = (_Float16*)alloc((size_t)128 * 128 * 2);
  _Float16* w1T   = (_Float16*)alloc((size_t)256 * 128 * 2);  // [M=256 x K=128]
  _Float16* w2T   = (_Float16*)alloc((size_t)128 * 256 * 2);  // [M=128 x K=256]
  float*    A     = (float*)alloc((size_t)N * 128 * 4);  // XL
  float*    B     = (float*)alloc((size_t)N * 128 * 4);  // XR
  float*    C     = (float*)alloc((size_t)N * 128 * 4);  // layer1 out / layer2 in
  float*    D     = (float*)alloc((size_t)N * 128 * 4);  // layer2 out
  float*    Hbuf  = (float*)alloc((size_t)N * 256 * 4);  // head hidden 1
  float*    eBuf  = (float*)alloc((size_t)Etot * 4);
  unsigned* segmx = (unsigned*)alloc((size_t)N * 4);
  float*    segsm = (float*)alloc((size_t)N * 4);
  float*    score = (float*)alloc((size_t)N * 4);
  float*    stats = (float*)alloc(512 * 4);
  float*    sm    = (float*)alloc(2 * 4);

  // ---- fp16 weight prep (deterministic each call) ----
  k_convert_T<<<(128 * 128 + 255) / 256, 256, 0, stream>>>(c1_wl, wlT1, 128, 128, 128);
  k_convert_T<<<(128 * 128 + 255) / 256, 256, 0, stream>>>(c1_wr, wrT1, 128, 128, 128);
  k_convert_T<<<(128 * 128 + 255) / 256, 256, 0, stream>>>(c2_wl, wlT2, 128, 128, 128);
  k_convert_T<<<(128 * 128 + 255) / 256, 256, 0, stream>>>(c2_wr, wrT2, 128, 128, 128);
  k_convert_T<<<(128 * 256 + 255) / 256, 256, 0, stream>>>(w1, w1T, 128, 256, 256); // top 128 rows
  k_convert_T<<<(256 * 128 + 255) / 256, 256, 0, stream>>>(w2, w2T, 256, 128, 128);

  // ---- global MLP (+ fold g @ w1[128:] + b1 into a bias vector) ----
  k_global_mlp<<<1, 128, 0, stream>>>(gx, wg1, bg1, wg2, bg2, gvec);
  k_gw1<<<1, 256, 0, stream>>>(gvec, w1, b1, gw1v);

  auto gemm = [&](const float* X, const _Float16* WT, const float* bias, float* Y,
                  int K, int M, int relu) {
    dim3 grid(M / 128, N / 16);
    k_gemm_wmma<<<grid, 256, 0, stream>>>(X, WT, bias, Y, K, M, relu);
  };

  const int eWave = (Etot + 7) / 8;       // wave-per-edge blocks (8 waves/block)
  auto gat = [&](const float* Xin, const _Float16* wlT, const _Float16* wrT,
                 const float* att, const float* bias,
                 const float* bng, const float* bnb, float* OUT) {
    gemm(Xin, wlT, nullptr, A, H, H, 0);   // XL = Xin @ wl
    gemm(Xin, wrT, nullptr, B, H, H, 0);   // XR = Xin @ wr
    k_fill_u32<<<(N + 255) / 256, 256, 0, stream>>>(segmx, ENC_NEG_INF, N);
    k_fill_u32<<<(N + 255) / 256, 256, 0, stream>>>((unsigned*)segsm, 0u, N);
    k_fill_u32<<<((N * H) + 255) / 256, 256, 0, stream>>>((unsigned*)OUT, 0u, N * H);
    k_edge_scores<<<eWave, 256, 0, stream>>>(ei, E, Etot, A, B, att, eBuf, segmx, H);
    k_edge_exp<<<(Etot + 255) / 256, 256, 0, stream>>>(ei, E, Etot, eBuf, segmx, segsm);
    k_edge_agg<<<eWave, 256, 0, stream>>>(ei, E, Etot, eBuf, segsm, A, OUT, H);
    long long n = (long long)N * H;
    k_bias_relu<<<(unsigned)((n + 255) / 256), 256, 0, stream>>>(OUT, bias, n, H);
    k_fill_u32<<<1, 256, 0, stream>>>((unsigned*)stats, 0u, 2 * H);
    k_bn_stats<<<240, H, 0, stream>>>(OUT, stats, N, H);
    k_bn_apply<<<(unsigned)((n + 255) / 256), 256, 0, stream>>>(OUT, stats, bng, bnb, n, N, H);
  };

  // ---- two GATv2 + BN layers ----
  gat(node, wlT1, wrT1, c1_at, c1_b, bn1_g, bn1_b, C);
  gat(C,    wlT2, wrT2, c2_at, c2_b, bn2_g, bn2_b, D);

  // ---- head: relu([x2, g] @ w1 + b1) == relu(x2 @ w1_top + gw1v) ----
  gemm(D, w1T, gw1v, Hbuf, 128, 256, 1);
  {
    long long n = (long long)N * 256;
    k_fill_u32<<<2, 256, 0, stream>>>((unsigned*)stats, 0u, 512);
    k_bn_stats<<<240, 256, 0, stream>>>(Hbuf, stats, N, 256);
    k_bn_apply<<<(unsigned)((n + 255) / 256), 256, 0, stream>>>(Hbuf, stats, fbn1g, fbn1b, n, N, 256);
  }
  gemm(Hbuf, w2T, b2, A, 256, 128, 1);
  {
    long long n = (long long)N * 128;
    k_fill_u32<<<1, 256, 0, stream>>>((unsigned*)stats, 0u, 256);
    k_bn_stats<<<240, 128, 0, stream>>>(A, stats, N, 128);
    k_bn_apply<<<(unsigned)((n + 255) / 256), 256, 0, stream>>>(A, stats, fbn2g, fbn2b, n, N, 128);
  }

  // ---- scores + softmax over nodes ----
  k_rowdot_relu<<<(N + 7) / 8, 256, 0, stream>>>(A, w3, b3, score, N, 128);
  k_softmax_reduce<<<1, 1024, 0, stream>>>(score, sm, N);
  k_softmax_write<<<(N + 255) / 256, 256, 0, stream>>>(score, sm, (float*)d_out, N);
}
